// SpatioTemporalTransformer_46445776339209
// MI455X (gfx1250) — compile-verified, matrix operand order checked by reading, not run on hardware
//
#include <hip/hip_runtime.h>
#include <hip/hip_bf16.h>
#include <math.h>

// ---------------- problem constants ----------------
#define N_NODE 10000
#define T_SEQ  64
#define H_DIM  128
#define N_HEAD 4
#define HD_DIM 32
#define L_LAY  2
#define S_BLK  3
#define E_EDGE 320000
#define O_OUT  12

typedef __attribute__((ext_vector_type(16))) _Float16 v16h;
typedef __attribute__((ext_vector_type(8)))  float    v8f;

// ---------------- WMMA fragment helpers (CDNA5 16x16x32 f16 layouts) ----------------
__device__ __forceinline__ int kmap(int e, int g) { return e + 8 * g + (e >= 8 ? 8 : 0); }

// Load fragment where storage is row-major [rows][stride] and we index (rbase+r, kbase+k).
// Serves A (rows = M) and B^T (rows = Nout, i.e. weight W[n][k]).
__device__ __forceinline__ v16h load_frag_rowK(const _Float16* p, int stride, int rbase, int kbase, int lane) {
    int g = lane >> 4, r = lane & 15;
    const _Float16* row = p + (size_t)(rbase + r) * stride + kbase;
    v16h f;
#pragma unroll
    for (int e = 0; e < 16; ++e) f[e] = row[kmap(e, g)];
    return f;
}

// Load B fragment from row-major [K][ncols] storage: B[k][n].
__device__ __forceinline__ v16h load_fragB_KN(const _Float16* p, int stride, int kbase, int nbase, int lane) {
    int g = lane >> 4, n = lane & 15;
    v16h f;
#pragma unroll
    for (int e = 0; e < 16; ++e) f[e] = p[(size_t)(kbase + kmap(e, g)) * stride + nbase + n];
    return f;
}

__device__ __forceinline__ v8f wmma_f16(v16h a, v16h b, v8f c) {
    return __builtin_amdgcn_wmma_f32_16x16x32_f16(false, a, false, b, (short)0, c, false, false);
}

// ---------------- float->f16 weight conversion ----------------
__global__ void cvt_f16_kernel(_Float16* __restrict__ dst, const float* __restrict__ src, int n) {
    int i = blockIdx.x * 256 + threadIdx.x;
    if (i < n) dst[i] = (_Float16)src[i];
}

__global__ void fill_int_kernel(int* __restrict__ p, int v, int n) {
    int i = blockIdx.x * 256 + threadIdx.x;
    if (i < n) p[i] = v;
}

// ---------------- temporal transformer: one workgroup per node ----------------
__device__ __forceinline__ void ln_rows(const float (*hb)[H_DIM], _Float16 (*z)[H_DIM],
                                        const float* w, const float* b,
                                        float* mu, float* rsig, int tid) {
    if (tid < T_SEQ) {
        float s = 0.f, s2 = 0.f;
        for (int c = 0; c < H_DIM; ++c) { float v = hb[tid][c]; s += v; s2 += v * v; }
        float m = s * (1.f / H_DIM);
        float var = s2 * (1.f / H_DIM) - m * m;
        mu[tid] = m;
        rsig[tid] = rsqrtf(var + 1e-5f);
    }
    __syncthreads();
    float wc = w[tid], bc = b[tid];
    for (int t = 0; t < T_SEQ; ++t)
        z[t][tid] = (_Float16)((hb[t][tid] - mu[t]) * rsig[t] * wc + bc);
    __syncthreads();
}

__global__ __launch_bounds__(128) void temporal_kernel(
    const float* __restrict__ x, const float* __restrict__ proj_w, const float* __restrict__ proj_b,
    const float* __restrict__ pos_emb,
    const _Float16* __restrict__ wqkv, const float* __restrict__ bqkv,
    const _Float16* __restrict__ wout, const float* __restrict__ bout,
    const _Float16* __restrict__ wff1, const float* __restrict__ bff1,
    const _Float16* __restrict__ wff2, const float* __restrict__ bff2,
    const float* __restrict__ ln1w, const float* __restrict__ ln1b,
    const float* __restrict__ ln2w, const float* __restrict__ ln2b,
    const float* __restrict__ nw, const float* __restrict__ nb,
    float* __restrict__ hmean, _Float16* __restrict__ h16)
{
    __shared__ float     hb[T_SEQ][H_DIM];      // 32 KB, persistent fp32 residual stream
    __shared__ _Float16  z16[T_SEQ][H_DIM];     // 16 KB, LN output / attn output (A operand)
    __shared__ _Float16  scratch[40960];        // 80 KB: qkv16(64x384) + att(4x64x64)  | reused: ff16(64x512)
    __shared__ float     mu[T_SEQ], rsig[T_SEQ];

    _Float16* qkv16  = scratch;                   // [64][384]
    _Float16* attbuf = scratch + T_SEQ * 384;     // [4][64][64]
    _Float16* ff16   = scratch;                   // [64][512]

    const int node = blockIdx.x;
    const int tid = threadIdx.x;
    const int lane = tid & 31;
    const int wv = tid >> 5;

    // ---- input projection + positional embedding: h[t][c] ----
    {
        float pw = proj_w[tid], pb = proj_b[tid];
        for (int t = 0; t < T_SEQ; ++t)
            hb[t][tid] = x[node * T_SEQ + t] * pw + pb + pos_emb[t * H_DIM + tid];
    }
    __syncthreads();

    for (int l = 0; l < L_LAY; ++l) {
        // ---- LN1 -> z16 ----
        ln_rows(hb, z16, ln1w + l * H_DIM, ln1b + l * H_DIM, mu, rsig, tid);

        // ---- qkv = z @ Wqkv^T + b  (M=64 -> m-tile per wave, N=384, K=128) ----
        {
            const _Float16* W = wqkv + (size_t)l * 384 * H_DIM;
            const float* B = bqkv + l * 384;
            for (int nt = 0; nt < 24; ++nt) {
                v8f acc = {};
                for (int kt = 0; kt < 4; ++kt) {
                    v16h a = load_frag_rowK(&z16[0][0], H_DIM, wv * 16, kt * 32, lane);
                    v16h bf = load_frag_rowK(W, H_DIM, nt * 16, kt * 32, lane);
                    acc = wmma_f16(a, bf, acc);
                }
                int g = lane >> 4, n = nt * 16 + (lane & 15);
                float bias = B[n];
#pragma unroll
                for (int v = 0; v < 8; ++v) {
                    int m = wv * 16 + v + 8 * g;
                    qkv16[m * 384 + n] = (_Float16)(acc[v] + bias);
                }
            }
        }
        __syncthreads();

        // ---- attention: wave = head. scores K=32 (=HD) -> one WMMA per tile ----
        {
            const int hh = wv;
            const int q0 = hh * 32, k0 = H_DIM + hh * 32;
            _Float16* att = attbuf + hh * T_SEQ * T_SEQ;
            const float scale = 0.17677669529663687f;  // 1/sqrt(32)
            for (int mt = 0; mt < 4; ++mt) {
                v8f sc[4];
                v16h a = load_frag_rowK(qkv16, 384, mt * 16, q0, lane);
#pragma unroll
                for (int nt = 0; nt < 4; ++nt) {
                    v8f z = {};
                    v16h bf = load_frag_rowK(qkv16, 384, nt * 16, k0, lane);  // B[k][n]=K[n][k]
                    sc[nt] = wmma_f16(a, bf, z);
                }
#pragma unroll
                for (int nt = 0; nt < 4; ++nt)
#pragma unroll
                    for (int v = 0; v < 8; ++v) sc[nt][v] *= scale;
                // row softmax: row m lives in one 16-lane half; reduce with xor masks 1..8
                int g = lane >> 4, nidx = lane & 15;
#pragma unroll
                for (int v = 0; v < 8; ++v) {
                    float mx = fmaxf(fmaxf(sc[0][v], sc[1][v]), fmaxf(sc[2][v], sc[3][v]));
                    for (int msk = 1; msk < 16; msk <<= 1) mx = fmaxf(mx, __shfl_xor(mx, msk, 32));
                    float p[4], sum = 0.f;
#pragma unroll
                    for (int nt = 0; nt < 4; ++nt) { p[nt] = __expf(sc[nt][v] - mx); sum += p[nt]; }
                    for (int msk = 1; msk < 16; msk <<= 1) sum += __shfl_xor(sum, msk, 32);
                    float inv = 1.f / sum;
                    int m = mt * 16 + v + 8 * g;
#pragma unroll
                    for (int nt = 0; nt < 4; ++nt)
                        att[m * 64 + nt * 16 + nidx] = (_Float16)(p[nt] * inv);
                }
            }
            // ---- o = att @ v  (M=64, N=32, K=64), write into z16 columns of this head ----
            for (int mt = 0; mt < 4; ++mt)
                for (int nt = 0; nt < 2; ++nt) {
                    v8f acc = {};
                    for (int kt = 0; kt < 2; ++kt) {
                        v16h a = load_frag_rowK(att, 64, mt * 16, kt * 32, lane);
                        v16h bf = load_fragB_KN(qkv16, 384, kt * 32, 2 * H_DIM + hh * 32 + nt * 16, lane);
                        acc = wmma_f16(a, bf, acc);
                    }
                    int g = lane >> 4, n = hh * 32 + nt * 16 + (lane & 15);
#pragma unroll
                    for (int v = 0; v < 8; ++v) {
                        int m = mt * 16 + v + 8 * g;
                        z16[m][n] = (_Float16)acc[v];
                    }
                }
        }
        __syncthreads();

        // ---- out-proj + residual: h += z16 @ Wout^T + b ----
        {
            const _Float16* W = wout + (size_t)l * H_DIM * H_DIM;
            for (int nt = 0; nt < 8; ++nt) {
                v8f acc = {};
                for (int kt = 0; kt < 4; ++kt) {
                    v16h a = load_frag_rowK(&z16[0][0], H_DIM, wv * 16, kt * 32, lane);
                    v16h bf = load_frag_rowK(W, H_DIM, nt * 16, kt * 32, lane);
                    acc = wmma_f16(a, bf, acc);
                }
                int g = lane >> 4, n = nt * 16 + (lane & 15);
                float bias = bout[l * H_DIM + n];
#pragma unroll
                for (int v = 0; v < 8; ++v) {
                    int m = wv * 16 + v + 8 * g;
                    hb[m][n] += acc[v] + bias;
                }
            }
        }
        __syncthreads();

        // ---- LN2 -> z16 ----
        ln_rows(hb, z16, ln2w + l * H_DIM, ln2b + l * H_DIM, mu, rsig, tid);

        // ---- ff1: relu(z16 @ Wff1^T + b) -> ff16 [64][512] ----
        {
            const _Float16* W = wff1 + (size_t)l * 512 * H_DIM;
            for (int nt = 0; nt < 32; ++nt) {
                v8f acc = {};
                for (int kt = 0; kt < 4; ++kt) {
                    v16h a = load_frag_rowK(&z16[0][0], H_DIM, wv * 16, kt * 32, lane);
                    v16h bf = load_frag_rowK(W, H_DIM, nt * 16, kt * 32, lane);
                    acc = wmma_f16(a, bf, acc);
                }
                int g = lane >> 4, n = nt * 16 + (lane & 15);
                float bias = bff1[l * 512 + n];
#pragma unroll
                for (int v = 0; v < 8; ++v) {
                    int m = wv * 16 + v + 8 * g;
                    float t = acc[v] + bias;
                    ff16[m * 512 + n] = (_Float16)(t > 0.f ? t : 0.f);
                }
            }
        }
        __syncthreads();

        // ---- ff2: h += ff16 @ Wff2^T + b  (K=512) ----
        {
            const _Float16* W = wff2 + (size_t)l * H_DIM * 512;
            for (int nt = 0; nt < 8; ++nt) {
                v8f acc = {};
                for (int kt = 0; kt < 16; ++kt) {
                    v16h a = load_frag_rowK(ff16, 512, wv * 16, kt * 32, lane);
                    v16h bf = load_frag_rowK(W, 512, nt * 16, kt * 32, lane);
                    acc = wmma_f16(a, bf, acc);
                }
                int g = lane >> 4, n = nt * 16 + (lane & 15);
                float bias = bff2[l * H_DIM + n];
#pragma unroll
                for (int v = 0; v < 8; ++v) {
                    int m = wv * 16 + v + 8 * g;
                    hb[m][n] += acc[v] + bias;
                }
            }
        }
        __syncthreads();
    }

    // ---- final LN + mean over T -> hmean[node], h16[node] ----
    if (tid < T_SEQ) {
        float s = 0.f, s2 = 0.f;
        for (int c = 0; c < H_DIM; ++c) { float v = hb[tid][c]; s += v; s2 += v * v; }
        float m = s * (1.f / H_DIM);
        float var = s2 * (1.f / H_DIM) - m * m;
        mu[tid] = m;
        rsig[tid] = rsqrtf(var + 1e-5f);
    }
    __syncthreads();
    {
        float s = 0.f;
        for (int t = 0; t < T_SEQ; ++t) s += (hb[t][tid] - mu[t]) * rsig[t];
        float val = s * (1.f / T_SEQ) * nw[tid] + nb[tid];
        hmean[(size_t)node * H_DIM + tid] = val;
        h16[(size_t)node * H_DIM + tid] = (_Float16)val;
    }
}

// ---------------- generic WMMA GEMM: out = act(A[MxK] @ B[NoutxK]^T + bias) ----------------
__global__ __launch_bounds__(128) void gemm_kernel(
    const _Float16* __restrict__ A, const _Float16* __restrict__ B, const float* __restrict__ bias,
    float* __restrict__ outF, _Float16* __restrict__ outH,
    int M, int K, int Ntiles, int Nstore, int act)
{
    int wv = threadIdx.x >> 5, lane = threadIdx.x & 31;
    int mbase = (blockIdx.x * 4 + wv) * 16;
    if (mbase >= M) return;
    for (int nt = 0; nt < Ntiles; ++nt) {
        v8f acc = {};
        for (int kt = 0; kt < K / 32; ++kt) {
            v16h a = load_frag_rowK(A, K, mbase, kt * 32, lane);
            v16h bf = load_frag_rowK(B, K, nt * 16, kt * 32, lane);
            acc = wmma_f16(a, bf, acc);
        }
        int g = lane >> 4, n = nt * 16 + (lane & 15);
        float bv = (bias != nullptr && n < Nstore) ? bias[n] : 0.f;
#pragma unroll
        for (int v = 0; v < 8; ++v) {
            int m = mbase + v + 8 * g;
            float val = acc[v] + bv;
            if (act) val = fmaxf(val, 0.f);
            if (n < Nstore) {
                size_t idx = (size_t)m * Nstore + n;
                if (outF) outF[idx] = val;
                if (outH) outH[idx] = (_Float16)val;
            }
        }
    }
}

// ---------------- GATv2 edge passes ----------------
__device__ __forceinline__ int fkey(float f) {
    int i = __float_as_int(f);
    return i >= 0 ? i : (i ^ 0x7fffffff);
}
__device__ __forceinline__ float funkey(int k) {
    return k >= 0 ? __int_as_float(k) : __int_as_float(k ^ 0x7fffffff);
}

__global__ void edge_alpha_kernel(const int* __restrict__ ei, const float* __restrict__ eattr,
                                  const float* __restrict__ we, const float* __restrict__ satt,
                                  const float* __restrict__ xl, const float* __restrict__ xr,
                                  float* __restrict__ alpha, int* __restrict__ mkey) {
    int e = blockIdx.x * 256 + threadIdx.x;
    if (e >= E_EDGE) return;
    int src = ei[e], dst = ei[E_EDGE + e];
    float ev = eattr[e];
    const float* xls = xl + (size_t)src * H_DIM;
    const float* xrd = xr + (size_t)dst * H_DIM;
#pragma unroll
    for (int h = 0; h < N_HEAD; ++h) {
        float a = 0.f;
        for (int d = 0; d < HD_DIM; ++d) {
            int c = h * HD_DIM + d;
            float t = xls[c] + xrd[c] + ev * we[c];
            t = t > 0.f ? t : 0.2f * t;
            a += t * satt[c];
        }
        alpha[(size_t)e * N_HEAD + h] = a;
        atomicMax(&mkey[dst * N_HEAD + h], fkey(a));
    }
}

__global__ void mfix_kernel(int* __restrict__ mk, int n) {
    int i = blockIdx.x * 256 + threadIdx.x;
    if (i >= n) return;
    float m = funkey(mk[i]);
    if (!isfinite(m)) m = 0.f;
    ((float*)mk)[i] = m;
}

__global__ void edge_exp_kernel(const int* __restrict__ ei, float* __restrict__ alpha,
                                const float* __restrict__ mfl, float* __restrict__ den) {
    int e = blockIdx.x * 256 + threadIdx.x;
    if (e >= E_EDGE) return;
    int dst = ei[E_EDGE + e];
#pragma unroll
    for (int h = 0; h < N_HEAD; ++h) {
        float v = __expf(alpha[(size_t)e * N_HEAD + h] - mfl[dst * N_HEAD + h]);
        alpha[(size_t)e * N_HEAD + h] = v;
        atomicAdd(&den[dst * N_HEAD + h], v);
    }
}

__global__ void edge_aggr_kernel(const int* __restrict__ ei, const float* __restrict__ ea,
                                 const float* __restrict__ den, const float* __restrict__ xl,
                                 float* __restrict__ outb) {
    int idx = blockIdx.x * 256 + threadIdx.x;
    if (idx >= E_EDGE * N_HEAD) return;
    int e = idx >> 2, h = idx & 3;
    int src = ei[e], dst = ei[E_EDGE + e];
    float w = ea[(size_t)e * N_HEAD + h] / (den[dst * N_HEAD + h] + 1e-16f);
    const float* xs = xl + (size_t)src * H_DIM + h * HD_DIM;
    float* od = outb + (size_t)dst * H_DIM + h * HD_DIM;
#pragma unroll
    for (int d = 0; d < HD_DIM; ++d) atomicAdd(&od[d], xs[d] * w);
}

__global__ void stats_kernel(const float* __restrict__ outb, const float* __restrict__ sbias,
                             float* __restrict__ stats, int n) {
    __shared__ float ss[256], ss2[256];
    int i = blockIdx.x * 256 + threadIdx.x;
    float v = 0.f;
    if (i < n) v = outb[i] + sbias[i & (H_DIM - 1)];
    ss[threadIdx.x] = v;
    ss2[threadIdx.x] = v * v;
    __syncthreads();
    for (int s = 128; s > 0; s >>= 1) {
        if ((int)threadIdx.x < s) {
            ss[threadIdx.x] += ss[threadIdx.x + s];
            ss2[threadIdx.x] += ss2[threadIdx.x + s];
        }
        __syncthreads();
    }
    if (threadIdx.x == 0) {
        atomicAdd(&stats[0], ss[0]);
        atomicAdd(&stats[1], ss2[0]);
    }
}

__global__ void gat_finalize_kernel(const float* __restrict__ outb, const float* __restrict__ sbias,
                                    const float* __restrict__ lnw, const float* __restrict__ lnb,
                                    const float* __restrict__ stats, float* __restrict__ hcur,
                                    _Float16* __restrict__ h16, int n) {
    int i = blockIdx.x * 256 + threadIdx.x;
    if (i >= n) return;
    float cnt = (float)n;
    float mean = stats[0] / cnt;
    float var = stats[1] / cnt - mean * mean;
    float sd = sqrtf(fmaxf(var, 0.f));
    int c = i & (H_DIM - 1);
    float g = (outb[i] + sbias[c] - mean) / (sd + 1e-5f) * lnw[c] + lnb[c];
    float el = g > 0.f ? g : (__expf(g) - 1.f);
    float nh = el + hcur[i];
    hcur[i] = nh;
    h16[i] = (_Float16)nh;
}

// ---------------- host launch ----------------
static inline size_t alignup(size_t v) { return (v + 255) & ~(size_t)255; }

extern "C" void kernel_launch(void* const* d_in, const int* in_sizes, int n_in,
                              void* d_out, int out_size, void* d_ws, size_t ws_size,
                              hipStream_t stream) {
    (void)in_sizes; (void)n_in; (void)out_size; (void)ws_size;
    const float* x        = (const float*)d_in[0];
    const int*   ei       = (const int*)d_in[1];
    const float* eattr    = (const float*)d_in[2];
    const float* proj_w   = (const float*)d_in[3];
    const float* proj_b   = (const float*)d_in[4];
    const float* pos_emb  = (const float*)d_in[5];
    const float* t_qkv_w  = (const float*)d_in[6];
    const float* t_qkv_b  = (const float*)d_in[7];
    const float* t_out_w  = (const float*)d_in[8];
    const float* t_out_b  = (const float*)d_in[9];
    const float* t_ln1_w  = (const float*)d_in[10];
    const float* t_ln1_b  = (const float*)d_in[11];
    const float* t_ln2_w  = (const float*)d_in[12];
    const float* t_ln2_b  = (const float*)d_in[13];
    const float* t_ff1_w  = (const float*)d_in[14];
    const float* t_ff1_b  = (const float*)d_in[15];
    const float* t_ff2_w  = (const float*)d_in[16];
    const float* t_ff2_b  = (const float*)d_in[17];
    const float* t_norm_w = (const float*)d_in[18];
    const float* t_norm_b = (const float*)d_in[19];
    const float* s_lin_l_w = (const float*)d_in[20];
    const float* s_lin_l_b = (const float*)d_in[21];
    const float* s_lin_r_w = (const float*)d_in[22];
    const float* s_lin_r_b = (const float*)d_in[23];
    const float* s_lin_e_w = (const float*)d_in[24];
    const float* s_att     = (const float*)d_in[25];
    const float* s_bias    = (const float*)d_in[26];
    const float* s_ln_w    = (const float*)d_in[27];
    const float* s_ln_b    = (const float*)d_in[28];
    const float* fc1_w     = (const float*)d_in[29];
    const float* fc1_b     = (const float*)d_in[30];
    const float* fc2_w     = (const float*)d_in[31];
    const float* fc2_b     = (const float*)d_in[32];

    // ---- workspace carve-up ----
    char* ws = (char*)d_ws;
    size_t off = 0;
    auto carve = [&](size_t bytes) { void* p = ws + off; off += alignup(bytes); return p; };
    _Float16* wqkv16 = (_Float16*)carve((size_t)L_LAY * 384 * 128 * 2);
    _Float16* wout16 = (_Float16*)carve((size_t)L_LAY * 128 * 128 * 2);
    _Float16* wff1_16 = (_Float16*)carve((size_t)L_LAY * 512 * 128 * 2);
    _Float16* wff2_16 = (_Float16*)carve((size_t)L_LAY * 128 * 512 * 2);
    _Float16* wlinl16 = (_Float16*)carve((size_t)S_BLK * 128 * 128 * 2);
    _Float16* wlinr16 = (_Float16*)carve((size_t)S_BLK * 128 * 128 * 2);
    _Float16* wfc1_16 = (_Float16*)carve((size_t)128 * 128 * 2);
    _Float16* wfc2_16 = (_Float16*)carve((size_t)16 * 128 * 2);  // padded 12->16 rows
    float*    hcur    = (float*)carve((size_t)N_NODE * H_DIM * 4);
    _Float16* h16     = (_Float16*)carve((size_t)N_NODE * H_DIM * 2);
    float*    xl      = (float*)carve((size_t)N_NODE * H_DIM * 4);
    float*    xr      = (float*)carve((size_t)N_NODE * H_DIM * 4);
    float*    alphab  = (float*)carve((size_t)E_EDGE * N_HEAD * 4);
    int*      mkey    = (int*)carve((size_t)N_NODE * N_HEAD * 4);
    float*    den     = (float*)carve((size_t)N_NODE * N_HEAD * 4);
    float*    outb    = (float*)carve((size_t)N_NODE * H_DIM * 4);
    _Float16* t1_16   = (_Float16*)carve((size_t)N_NODE * H_DIM * 2);
    float*    stats   = (float*)carve(256);

    auto cvt = [&](_Float16* dst, const float* src, int n) {
        cvt_f16_kernel<<<(n + 255) / 256, 256, 0, stream>>>(dst, src, n);
    };

    // ---- weight prep (f32 -> f16) ----
    cvt(wqkv16, t_qkv_w, L_LAY * 384 * 128);
    cvt(wout16, t_out_w, L_LAY * 128 * 128);
    cvt(wff1_16, t_ff1_w, L_LAY * 512 * 128);
    cvt(wff2_16, t_ff2_w, L_LAY * 128 * 512);
    cvt(wlinl16, s_lin_l_w, S_BLK * 128 * 128);
    cvt(wlinr16, s_lin_r_w, S_BLK * 128 * 128);
    cvt(wfc1_16, fc1_w, 128 * 128);
    hipMemsetAsync(wfc2_16, 0, (size_t)16 * 128 * 2, stream);
    cvt(wfc2_16, fc2_w, O_OUT * 128);

    // ---- phase 1: temporal transformer (one block per node) ----
    temporal_kernel<<<N_NODE, 128, 0, stream>>>(
        x, proj_w, proj_b, pos_emb,
        wqkv16, t_qkv_b, wout16, t_out_b, wff1_16, t_ff1_b, wff2_16, t_ff2_b,
        t_ln1_w, t_ln1_b, t_ln2_w, t_ln2_b, t_norm_w, t_norm_b,
        hcur, h16);

    const int gemmBlocks = (N_NODE / 16 + 3) / 4;  // 625 m-tiles, 4 waves/block
    const int NEG_INF_KEY = (int)0x807FFFFF;       // fkey(-inf)

    // ---- phase 2: stacked GATv2 blocks ----
    for (int s = 0; s < S_BLK; ++s) {
        gemm_kernel<<<gemmBlocks, 128, 0, stream>>>(
            h16, wlinl16 + (size_t)s * 128 * 128, s_lin_l_b + s * 128,
            xl, nullptr, N_NODE, 128, 8, 128, 0);
        gemm_kernel<<<gemmBlocks, 128, 0, stream>>>(
            h16, wlinr16 + (size_t)s * 128 * 128, s_lin_r_b + s * 128,
            xr, nullptr, N_NODE, 128, 8, 128, 0);

        hipMemsetAsync(den, 0, (size_t)N_NODE * N_HEAD * 4, stream);
        hipMemsetAsync(outb, 0, (size_t)N_NODE * H_DIM * 4, stream);
        hipMemsetAsync(stats, 0, 8, stream);
        fill_int_kernel<<<(N_NODE * N_HEAD + 255) / 256, 256, 0, stream>>>(mkey, NEG_INF_KEY, N_NODE * N_HEAD);

        edge_alpha_kernel<<<(E_EDGE + 255) / 256, 256, 0, stream>>>(
            ei, eattr, s_lin_e_w + s * 128, s_att + s * 128, xl, xr, alphab, mkey);
        mfix_kernel<<<(N_NODE * N_HEAD + 255) / 256, 256, 0, stream>>>(mkey, N_NODE * N_HEAD);
        edge_exp_kernel<<<(E_EDGE + 255) / 256, 256, 0, stream>>>(ei, alphab, (const float*)mkey, den);
        edge_aggr_kernel<<<(E_EDGE * N_HEAD + 255) / 256, 256, 0, stream>>>(ei, alphab, den, xl, outb);

        stats_kernel<<<(N_NODE * H_DIM + 255) / 256, 256, 0, stream>>>(
            outb, s_bias + s * 128, stats, N_NODE * H_DIM);
        gat_finalize_kernel<<<(N_NODE * H_DIM + 255) / 256, 256, 0, stream>>>(
            outb, s_bias + s * 128, s_ln_w + s * 128, s_ln_b + s * 128,
            stats, hcur, h16, N_NODE * H_DIM);
    }

    // ---- phase 3: FC head ----
    gemm_kernel<<<gemmBlocks, 128, 0, stream>>>(
        h16, wfc1_16, fc1_b, nullptr, t1_16, N_NODE, 128, 8, 128, 1);
    gemm_kernel<<<gemmBlocks, 128, 0, stream>>>(
        t1_16, wfc2_16, fc2_b, (float*)d_out, nullptr, N_NODE, 128, 1, O_OUT, 0);
}
